// HierarchicalTargetAttention_75746043232942
// MI455X (gfx1250) — matrix operand order repaired and checked by reading, not compiled, not run on hardware
//
#include <hip/hip_runtime.h>

// ---------- WMMA types (CDNA5 gfx1250, wave32) ----------
typedef __attribute__((ext_vector_type(16))) __bf16 bf16x16;
typedef __attribute__((ext_vector_type(8)))  float  f32x8;

struct __align__(16) U4 { unsigned int x, y, z, w; };

union ABFrag {
  bf16x16 v;
  U4 q[2];
};

static __device__ __forceinline__ f32x8 wmma_bf16(bf16x16 a, bf16x16 b, f32x8 c) {
  // D = A(16x32 bf16) * B(32x16 bf16) + C(16x16 f32)
  return __builtin_amdgcn_wmma_f32_16x16x32_bf16(false, a, false, b, (short)0, c, false, false);
}

// A fragment: lane (m=lane&15, h=lane>>4) holds A[m][k + h*8 .. +7] and A[m][k+16 + h*8 .. +7]
static __device__ __forceinline__ void load_a(ABFrag& f, const __bf16* __restrict__ p) {
  f.q[0] = *(const U4*)p;
  f.q[1] = *(const U4*)(p + 16);
}
// B fragment (Bt row-major, k contiguous): lane (n=lane&15, h) holds Bt[n][k + h*8 .. +7] and Bt[n][k+16 + h*8 .. +7]
static __device__ __forceinline__ void load_b(ABFrag& f, const __bf16* __restrict__ p) {
  f.q[0] = *(const U4*)p;
  f.q[1] = *(const U4*)(p + 16);
}

// ---------- elementwise f32 -> bf16 ----------
__global__ void cvt_f32_bf16(const float* __restrict__ in, __bf16* __restrict__ out, int n) {
  int stride = gridDim.x * blockDim.x;
  for (int i = blockIdx.x * blockDim.x + threadIdx.x; i < n; i += stride)
    out[i] = (__bf16)in[i];
}

// ---------- H [b][f][l] f32 -> Ht [b][l][f] bf16 (LDS tiled transpose) ----------
__global__ __launch_bounds__(256)
void transpose_h(const float* __restrict__ H, __bf16* __restrict__ Ht, int d, int L) {
  __shared__ float tile[32][33];
  int b = blockIdx.z;
  int l0 = blockIdx.x * 32, f0 = blockIdx.y * 32;
  const float* Hb = H + (size_t)b * d * L;
  __bf16* Htb = Ht + (size_t)b * L * d;
  int x = threadIdx.x, y = threadIdx.y;  // 32 x 8
  #pragma unroll
  for (int j = 0; j < 32; j += 8)
    tile[y + j][x] = Hb[(size_t)(f0 + y + j) * L + l0 + x];
  __syncthreads();
  #pragma unroll
  for (int j = 0; j < 32; j += 8)
    Htb[(size_t)(l0 + y + j) * d + f0 + x] = (__bf16)tile[x][y + j];
}

// ---------- generic NT GEMM with bias + ELU, bf16 out ----------
// D[m][n] = ELU( sum_k A[m][k]*Bt[n][k] + bias ), block tile 64x64, 4 waves of 32x32
__global__ __launch_bounds__(128)
void gemm_nt_elu(const __bf16* __restrict__ A, long asb, int lda,
                 const __bf16* __restrict__ Bt, long bsb, int ldb,
                 const float* __restrict__ bias, int bias_on_m,
                 __bf16* __restrict__ D, long dsb, int ldd,
                 int M, int N, int Kd) {
  int b = blockIdx.z;
  A  += (size_t)b * asb;
  Bt += (size_t)b * bsb;
  D  += (size_t)b * dsb;
  int lane = threadIdx.x & 31;
  int wave = threadIdx.x >> 5;               // 4 waves: 2x2
  int wm = wave >> 1, wn = wave & 1;
  int m0 = blockIdx.y * 64 + wm * 32;
  int n0 = blockIdx.x * 64 + wn * 32;
  int lr = lane & 15, lh = lane >> 4;
  f32x8 acc[2][2] = {};
  for (int k = 0; k < Kd; k += 32) {
    ABFrag a[2], bf[2];
    #pragma unroll
    for (int i = 0; i < 2; ++i) {
      const __bf16* pa = A + (size_t)(m0 + i * 16 + lr) * lda + k + lh * 8;
      __builtin_prefetch(pa + 32, 0, 1);
      load_a(a[i], pa);
    }
    #pragma unroll
    for (int j = 0; j < 2; ++j) {
      const __bf16* pb = Bt + (size_t)(n0 + j * 16 + lr) * ldb + k + lh * 8;
      __builtin_prefetch(pb + 32, 0, 1);
      load_b(bf[j], pb);
    }
    #pragma unroll
    for (int i = 0; i < 2; ++i)
      #pragma unroll
      for (int j = 0; j < 2; ++j)
        acc[i][j] = wmma_bf16(a[i].v, bf[j].v, acc[i][j]);
  }
  #pragma unroll
  for (int i = 0; i < 2; ++i) {
    #pragma unroll
    for (int j = 0; j < 2; ++j) {
      int col = n0 + j * 16 + lr;
      #pragma unroll
      for (int r = 0; r < 8; ++r) {
        int row = m0 + i * 16 + lh * 8 + r;
        float v = acc[i][j][r] + (bias_on_m ? bias[row] : bias[col]);
        v = v > 0.0f ? v : (__expf(v) - 1.0f);  // ELU(alpha=1)
        if (row < M && col < N)
          D[(size_t)row * ldd + col] = (__bf16)v;
      }
    }
  }
}

// ---------- attention logits: E[c][l] = scale * sum_e Q[c][e]*K[l][e] ----------
// block tile 32x256, 8 waves of 16x64 (A-fragment rows clamped to keep EXEC uniform)
__global__ __launch_bounds__(256)
void attn_logits(const __bf16* __restrict__ Q, long qsb,
                 const __bf16* __restrict__ K, long ksb,
                 float* __restrict__ E, long esb,
                 int M, int L, int d, float scale) {
  int b = blockIdx.z;
  Q += (size_t)b * qsb;
  K += (size_t)b * ksb;
  E += (size_t)b * esb;
  int lane = threadIdx.x & 31;
  int wave = threadIdx.x >> 5;               // 8 waves: 2x4
  int wm = wave >> 2, wn = wave & 3;
  int m0 = blockIdx.x * 32 + wm * 16;
  int n0 = blockIdx.y * 256 + wn * 64;
  int lr = lane & 15, lh = lane >> 4;
  int rowA = m0 + lr;
  if (rowA > M - 1) rowA = M - 1;            // clamp: stay in-bounds, stores are guarded
  f32x8 acc[4] = {};
  for (int k = 0; k < d; k += 32) {
    ABFrag a, bf[4];
    const __bf16* pa = Q + (size_t)rowA * d + k + lh * 8;
    __builtin_prefetch(pa + 32, 0, 1);
    load_a(a, pa);
    #pragma unroll
    for (int j = 0; j < 4; ++j) {
      const __bf16* pb = K + (size_t)(n0 + j * 16 + lr) * d + k + lh * 8;
      __builtin_prefetch(pb + 32, 0, 1);
      load_b(bf[j], pb);
    }
    #pragma unroll
    for (int j = 0; j < 4; ++j)
      acc[j] = wmma_bf16(a.v, bf[j].v, acc[j]);
  }
  #pragma unroll
  for (int j = 0; j < 4; ++j) {
    int col = n0 + j * 16 + lr;
    #pragma unroll
    for (int r = 0; r < 8; ++r) {
      int row = m0 + lh * 8 + r;
      if (row < M)
        E[(size_t)row * L + col] = acc[j][r] * scale;
    }
  }
}

// ---------- in-place row softmax; one block per row, row cached in LDS ----------
__global__ __launch_bounds__(256)
void softmax_row(float* __restrict__ E, int L) {
  __shared__ float row[4096];
  __shared__ float red[256];
  size_t base = (size_t)blockIdx.x * L;
  int t = threadIdx.x;
  float lmax = -3.4e38f;
  for (int i = t; i < L; i += 256) {
    float v = E[base + i];
    row[i] = v;
    lmax = fmaxf(lmax, v);
  }
  red[t] = lmax;
  __syncthreads();
  for (int s = 128; s > 0; s >>= 1) {
    if (t < s) red[t] = fmaxf(red[t], red[t + s]);
    __syncthreads();
  }
  float m = red[0];
  __syncthreads();
  float lsum = 0.0f;
  for (int i = t; i < L; i += 256) {
    float p = __expf(row[i] - m);
    row[i] = p;
    lsum += p;
  }
  red[t] = lsum;
  __syncthreads();
  for (int s = 128; s > 0; s >>= 1) {
    if (t < s) red[t] += red[t + s];
    __syncthreads();
  }
  float rinv = 1.0f / red[0];
  for (int i = t; i < L; i += 256) E[base + i] = row[i] * rinv;
}

// ---------- C[c][e] = sum_l P[c][l] * Vt[e][l]  (P f32 -> bf16 on the fly) ----------
// Full-width N: block tile 32 x 512 (all of d), 8 waves as 2(m) x 4(n),
// wave tile 16 x 128 -> 8 accumulators. Each P row is streamed from HBM once.
__global__ __launch_bounds__(256)
void av_gemm(const float* __restrict__ P, long psb,
             const __bf16* __restrict__ Vt, long vsb,
             float* __restrict__ C, long csb,
             int M, int L, int d) {
  int b = blockIdx.z;
  P  += (size_t)b * psb;
  Vt += (size_t)b * vsb;
  C  += (size_t)b * csb;
  int lane = threadIdx.x & 31;
  int wave = threadIdx.x >> 5;               // 8 waves: 2x4
  int wm = wave >> 2, wn = wave & 3;
  int m0 = blockIdx.x * 32 + wm * 16;
  int n0 = wn * 128;                         // N covered fully by the block
  int lr = lane & 15, lh = lane >> 4;
  int rowA = m0 + lr;
  if (rowA > M - 1) rowA = M - 1;
  f32x8 acc[8] = {};
  for (int k = 0; k < L; k += 32) {
    ABFrag a, bf[8];
    const float* pa = P + (size_t)rowA * L + k + lh * 8;
    __builtin_prefetch(pa + 32, 0, 1);
    union { U4 u[4]; float f[16]; } aw;
    aw.u[0] = *(const U4*)pa;
    aw.u[1] = *(const U4*)(pa + 4);
    aw.u[2] = *(const U4*)(pa + 16);
    aw.u[3] = *(const U4*)(pa + 20);
    #pragma unroll
    for (int i = 0; i < 16; ++i) a.v[i] = (__bf16)aw.f[i];
    #pragma unroll
    for (int j = 0; j < 8; ++j) {
      const __bf16* pb = Vt + (size_t)(n0 + j * 16 + lr) * L + k + lh * 8;
      __builtin_prefetch(pb + 32, 0, 1);
      load_b(bf[j], pb);
    }
    #pragma unroll
    for (int j = 0; j < 8; ++j)
      acc[j] = wmma_bf16(a.v, bf[j].v, acc[j]);
  }
  #pragma unroll
  for (int j = 0; j < 8; ++j) {
    int col = n0 + j * 16 + lr;
    #pragma unroll
    for (int r = 0; r < 8; ++r) {
      int row = m0 + lh * 8 + r;
      if (row < M)
        C[(size_t)row * d + col] = acc[j][r];
    }
  }
}

// ---------- Q2b[b][n][e] = bf16( Q2[n][e] + C1[b][map[n]][e] ) ----------
__global__ void build_q2b(const float* __restrict__ Q2, const float* __restrict__ C1,
                          const int* __restrict__ map, __bf16* __restrict__ Q2b,
                          int B, int Lbl, int d, int LblPad, int C1n) {
  size_t n = (size_t)B * Lbl * d;
  size_t stride = (size_t)gridDim.x * blockDim.x;
  for (size_t i = (size_t)blockIdx.x * blockDim.x + threadIdx.x; i < n; i += stride) {
    int e = (int)(i % d);
    size_t t = i / d;
    int nn = (int)(t % Lbl);
    int b = (int)(t / Lbl);
    int cat = map[nn];
    float v = Q2[(size_t)nn * d + e] + C1[((size_t)b * C1n + cat) * d + e];
    Q2b[((size_t)b * LblPad + nn) * d + e] = (__bf16)v;
  }
}

extern "C" void kernel_launch(void* const* d_in, const int* in_sizes, int n_in,
                              void* d_out, int out_size, void* d_ws, size_t ws_size,
                              hipStream_t stream) {
  (void)in_sizes; (void)n_in; (void)out_size; (void)ws_size;
  const float* H  = (const float*)d_in[0];
  const float* Wk = (const float*)d_in[1];
  const float* bk = (const float*)d_in[2];
  const float* Wv = (const float*)d_in[3];
  const float* bv = (const float*)d_in[4];
  const float* Q1 = (const float*)d_in[5];
  const float* Q2 = (const float*)d_in[6];
  const int*  map = (const int*)d_in[7];

  const int B = 4, d = 512, L = 4096;
  const int C1n = 1168, C1p = 1184;     // padded to multiple of 32
  const int LBL = 8929, LBLp = 8960;    // padded to multiple of 32

  char* ws = (char*)d_ws;
  size_t off = 0;
  auto alloc = [&](size_t bytes) -> void* {
    void* p = ws + off;
    off += (bytes + 255) & ~(size_t)255;
    return p;
  };
  __bf16* Ht  = (__bf16*)alloc((size_t)B * L * d * 2);
  __bf16* Wkb = (__bf16*)alloc((size_t)d * d * 2);
  __bf16* Wvb = (__bf16*)alloc((size_t)d * d * 2);
  __bf16* Q1b = (__bf16*)alloc((size_t)C1p * d * 2);
  __bf16* Kb  = (__bf16*)alloc((size_t)B * L * d * 2);   // K[b][l][e]
  __bf16* Vtb = (__bf16*)alloc((size_t)B * d * L * 2);   // Vt[b][e][l]
  float*  E1  = (float*) alloc((size_t)B * C1n * L * 4);
  float*  C1b = (float*) alloc((size_t)B * C1n * d * 4);
  __bf16* Q2b = (__bf16*)alloc((size_t)B * LBLp * d * 2);

  float* C2out = (float*)d_out;                          // [B][LBL][d]
  float* A2out = C2out + (size_t)B * LBL * d;            // [B][LBL][L]

  const float rscale = 0.04419417382415922f;             // 1/sqrt(512)

  // 1) bf16 conversions of small operands
  cvt_f32_bf16<<<256, 256, 0, stream>>>(Wk, Wkb, d * d);
  cvt_f32_bf16<<<256, 256, 0, stream>>>(Wv, Wvb, d * d);
  cvt_f32_bf16<<<256, 256, 0, stream>>>(Q1, Q1b, C1n * d);

  // 2) H -> Ht (bf16, transposed)
  transpose_h<<<dim3(L / 32, d / 32, B), dim3(32, 8), 0, stream>>>(H, Ht, d, L);

  // 3) K[b][l][e] = ELU(Ht . Wk^T + bk)   (M=L, N=d, K=d)
  gemm_nt_elu<<<dim3(d / 64, L / 64, B), 128, 0, stream>>>(
      Ht, (long)L * d, d, Wkb, 0, d, bk, /*bias_on_m=*/0,
      Kb, (long)L * d, d, L, d, d);
  // 4) Vt[b][e][l] = ELU(Wv . Ht^T + bv)  (M=d, N=L, K=d)
  gemm_nt_elu<<<dim3(L / 64, d / 64, B), 128, 0, stream>>>(
      Wvb, 0, d, Ht, (long)L * d, d, bv, /*bias_on_m=*/1,
      Vtb, (long)d * L, L, d, L, d);

  // 5) Stage 1: E1 = Q1.K^T/scale ; softmax ; C1 = A1.V
  attn_logits<<<dim3((C1n + 31) / 32, L / 256, B), 256, 0, stream>>>(
      Q1b, 0, Kb, (long)L * d, E1, (long)C1n * L, C1n, L, d, rscale);
  softmax_row<<<B * C1n, 256, 0, stream>>>(E1, L);
  av_gemm<<<dim3((C1n + 31) / 32, 1, B), 256, 0, stream>>>(
      E1, (long)C1n * L, Vtb, (long)d * L, C1b, (long)C1n * d, C1n, L, d);

  // 6) Q2b = Q2 + C1[map]
  build_q2b<<<1024, 256, 0, stream>>>(Q2, C1b, map, Q2b, B, LBL, d, LBLp, C1n);

  // 7) Stage 2: A2 (output) and C2 (output)
  attn_logits<<<dim3((LBL + 31) / 32, L / 256, B), 256, 0, stream>>>(
      Q2b, (long)LBLp * d, Kb, (long)L * d, A2out, (long)LBL * L, LBL, L, d, rscale);
  softmax_row<<<B * LBL, 256, 0, stream>>>(A2out, L);
  av_gemm<<<dim3((LBL + 31) / 32, 1, B), 256, 0, stream>>>(
      A2out, (long)LBL * L, Vtb, (long)d * L, C2out, (long)LBL * d, LBL, L, d);
}
